// MoEMlp_6433861009665
// MI455X (gfx1250) — compile-verified, matrix operand order checked by reading, not run on hardware
//
#include <hip/hip_runtime.h>
#include <hip/hip_bf16.h>
#include <math.h>

// ---------------------------------------------------------------------------
// MoE MLP for MI455X (gfx1250, wave32, WMMA)
//   h   = GELU(x @ W1 + b1)              M=32768 K=1024 N=4096
//   out = [ h @ W2 + b2 | h @ We[idx[b]] + be[idx[b]] ]   M=32768 K=4096 N=1024
// v_wmma_f32_16x16x32_bf16, f32 accumulate. Intermediate h in bf16 in d_ws.
// 128x128 block tile, 8 waves (4M x 2N), 32x64 wave tile (2x4 WMMA tiles),
// double-buffered LDS; kernel-2 A-tile staged with GLOBAL_LOAD_ASYNC_TO_LDS.
// ---------------------------------------------------------------------------

typedef __attribute__((ext_vector_type(16))) __bf16 v16bf;
typedef __attribute__((ext_vector_type(8)))  float  v8f;

#define BM 128
#define BN 128
#define BK 32
#define ROWP 40            // padded LDS row in bf16 elems (80 B, 16B-aligned, bank-safe)
#define NT 256

#if __has_builtin(__builtin_amdgcn_global_load_async_to_lds_b128)
#define HAVE_ASYNC_LDS 1
#else
#define HAVE_ASYNC_LDS 0
#endif

#if HAVE_ASYNC_LDS
typedef int v4i_ __attribute__((__vector_size__(16)));
typedef __attribute__((address_space(1))) v4i_* gbl_v4i_p;   // __device__ int4*
typedef __attribute__((address_space(3))) v4i_* lds_v4i_p;   // __shared__ int4*
#endif

__device__ __forceinline__ unsigned bf16_bits(float f) {
    unsigned u = __float_as_uint(f);
    unsigned r = u + 0x7FFFu + ((u >> 16) & 1u);
    return (r >> 16) & 0xFFFFu;
}
__device__ __forceinline__ unsigned pack_bf16x2(float lo, float hi) {
    return (bf16_bits(hi) << 16) | bf16_bits(lo);
}
__device__ __forceinline__ float gelu_exact(float x) {
    return 0.5f * x * (1.0f + erff(x * 0.70710678118654752f));
}

// 16-byte global -> LDS copy: async DMA path on gfx1250 (ASYNCcnt), else flat copy.
__device__ __forceinline__ void copy16_g2l(const unsigned short* g, unsigned short* l) {
#if HAVE_ASYNC_LDS
    __builtin_amdgcn_global_load_async_to_lds_b128(
        (gbl_v4i_p)(void*)g, (lds_v4i_p)(void*)l, 0, 0);
#else
    *(uint4*)l = *(const uint4*)g;
#endif
}
__device__ __forceinline__ void wait_async_copies() {
#if HAVE_ASYNC_LDS
#if __has_builtin(__builtin_amdgcn_s_wait_asynccnt)
    __builtin_amdgcn_s_wait_asynccnt(0);
#else
    asm volatile("s_wait_asynccnt 0x0" ::: "memory");
#endif
#endif
}

union FragU { v16bf v; uint4 q[2]; };

// 16x32 bf16 operand fragment from a padded LDS tile (A-layout; B uses the
// same loader on a transposed tile, lane = column).
__device__ __forceinline__ v16bf load_frag(const unsigned short* smem, int rowBase, int lane) {
    int r    = rowBase + (lane & 15);
    int half = lane >> 4;
    const char* p = (const char*)(smem + r * ROWP) + half * 16;
    FragU f;
    f.q[0] = *(const uint4*)(p);
    f.q[1] = *(const uint4*)(p + 32);
    return f.v;
}

#define WMMA_BF16(A, B, C) \
    __builtin_amdgcn_wmma_f32_16x16x32_bf16(false, (A), false, (B), (short)0, (C), false, false)

// ---------------------------------------------------------------------------
// Kernel 1: h = GELU(x @ W1 + b1), h stored as bf16
// ---------------------------------------------------------------------------
__global__ __launch_bounds__(NT)
void fc1_gelu_kernel(const float* __restrict__ X,    // [32768][1024] f32
                     const float* __restrict__ W1,   // [1024][4096]  f32
                     const float* __restrict__ B1,   // [4096]        f32
                     unsigned short* __restrict__ H) // [32768][4096] bf16
{
    __shared__ __align__(16) unsigned short As[2][BM * ROWP];
    __shared__ __align__(16) unsigned short Bs[2][BN * ROWP];

    const int K = 1024, LDA = 1024, LDB = 4096, LDH = 4096;
    const int tid  = threadIdx.x;
    const int lane = tid & 31;
    const int wave = tid >> 5;
    const int wm = (wave & 3) * 32;     // 4 waves along M, 32 rows each
    const int wn = (wave >> 2) * 64;    // 2 waves along N, 64 cols each
    const int mBase = blockIdx.y * BM;
    const int nBase = blockIdx.x * BN;

    v8f zero = {};
    v8f acc[2][4];
#pragma unroll
    for (int i = 0; i < 2; ++i)
#pragma unroll
        for (int j = 0; j < 4; ++j) acc[i][j] = zero;

    const int nK = K / BK;

    // ---- stage buffer 0 (f32 -> bf16 conversion in flight)
    {
        unsigned* Au = (unsigned*)As[0];
#pragma unroll
        for (int i = 0; i < 8; ++i) {
            int p = tid + i * NT;                    // 0..2047 (128 rows x 16 pairs)
            int row = p >> 4, kp = p & 15;
            float2 v = *(const float2*)(X + (size_t)(mBase + row) * LDA + kp * 2);
            Au[row * (ROWP / 2) + kp] = pack_bf16x2(v.x, v.y);
        }
#pragma unroll
        for (int i = 0; i < 16; ++i) {
            int e = tid + i * NT;                    // 0..4095 (32 k x 128 n)
            int k = e >> 7, n = e & 127;
            float w = W1[(size_t)k * LDB + (nBase + n)];
            Bs[0][n * ROWP + k] = (unsigned short)bf16_bits(w);
        }
    }
    __syncthreads();

    for (int kt = 0; kt < nK; ++kt) {
        const int cur = kt & 1;
        // ---- stage next buffer while computing current
        if (kt + 1 < nK) {
            const int k0 = (kt + 1) * BK;
            unsigned* Au = (unsigned*)As[cur ^ 1];
            unsigned short* Bp = Bs[cur ^ 1];
#pragma unroll
            for (int i = 0; i < 8; ++i) {
                int p = tid + i * NT;
                int row = p >> 4, kp = p & 15;
                float2 v = *(const float2*)(X + (size_t)(mBase + row) * LDA + (k0 + kp * 2));
                Au[row * (ROWP / 2) + kp] = pack_bf16x2(v.x, v.y);
            }
#pragma unroll
            for (int i = 0; i < 16; ++i) {
                int e = tid + i * NT;
                int k = e >> 7, n = e & 127;
                float w = W1[(size_t)(k0 + k) * LDB + (nBase + n)];
                Bp[n * ROWP + k] = (unsigned short)bf16_bits(w);
            }
        }
        // ---- compute on current buffer: 2 A frags x 4 B frags = 8 WMMA
        const unsigned short* Ap = As[cur];
        const unsigned short* Bp = Bs[cur];
        v16bf a0 = load_frag(Ap, wm,      lane);
        v16bf a1 = load_frag(Ap, wm + 16, lane);
        v16bf b0 = load_frag(Bp, wn,      lane);
        v16bf b1 = load_frag(Bp, wn + 16, lane);
        v16bf b2 = load_frag(Bp, wn + 32, lane);
        v16bf b3 = load_frag(Bp, wn + 48, lane);
        acc[0][0] = WMMA_BF16(a0, b0, acc[0][0]);
        acc[0][1] = WMMA_BF16(a0, b1, acc[0][1]);
        acc[0][2] = WMMA_BF16(a0, b2, acc[0][2]);
        acc[0][3] = WMMA_BF16(a0, b3, acc[0][3]);
        acc[1][0] = WMMA_BF16(a1, b0, acc[1][0]);
        acc[1][1] = WMMA_BF16(a1, b1, acc[1][1]);
        acc[1][2] = WMMA_BF16(a1, b2, acc[1][2]);
        acc[1][3] = WMMA_BF16(a1, b3, acc[1][3]);
        __syncthreads();
    }

    // ---- epilogue: bias + exact GELU, store bf16
    const int rHalf = (lane >> 4) * 8;
    const int nLoc  = (lane & 15);
#pragma unroll
    for (int i = 0; i < 2; ++i) {
#pragma unroll
        for (int j = 0; j < 4; ++j) {
            int col  = nBase + wn + j * 16 + nLoc;
            float bv = B1[col];
            int rowB = mBase + wm + i * 16 + rHalf;
#pragma unroll
            for (int r = 0; r < 8; ++r) {
                float v = gelu_exact(acc[i][j][r] + bv);
                H[(size_t)(rowB + r) * LDH + col] = (unsigned short)bf16_bits(v);
            }
        }
    }
}

// ---------------------------------------------------------------------------
// Kernel 2: out = [ h@W2 + b2 | h@We[idx[b]] + be[idx[b]] ]
// BN=128: column blocks 0..5 use W2 (768 cols), 6..7 use the expert (256).
// A 128-row M tile lies inside one batch b (2048 % 128 == 0) -> uniform block.
// A tile (bf16 copy) staged with async global->LDS DMA on the ASYNCcnt path.
// ---------------------------------------------------------------------------
__global__ __launch_bounds__(NT)
void fc2_moe_kernel(const unsigned short* __restrict__ H,  // [32768][4096] bf16
                    const float* __restrict__ W2,          // [4096][768]
                    const float* __restrict__ B2,          // [768]
                    const float* __restrict__ We,          // [8][4096][256]
                    const float* __restrict__ Be,          // [8][256]
                    const int*   __restrict__ IDX,         // [16]
                    float* __restrict__ OUT)               // [32768][1024]
{
    __shared__ __align__(16) unsigned short As[2][BM * ROWP];
    __shared__ __align__(16) unsigned short Bs[2][BN * ROWP];

    const int K = 4096, LDH = 4096, LDO = 1024;
    const int tid  = threadIdx.x;
    const int lane = tid & 31;
    const int wave = tid >> 5;
    const int wm = (wave & 3) * 32;
    const int wn = (wave >> 2) * 64;
    const int mBase = blockIdx.y * BM;
    const int nBase = blockIdx.x * BN;
    const int b = mBase >> 11;                 // 2048 rows per batch element

    const float* wptr; const float* bptr; int ldb; int cb;
    if (nBase < 768) {
        wptr = W2; bptr = B2; ldb = 768; cb = nBase;
    } else {
        int e = IDX[b];
        wptr = We + (size_t)e * 4096 * 256;
        bptr = Be + (size_t)e * 256;
        ldb = 256; cb = nBase - 768;
    }

    v8f zero = {};
    v8f acc[2][4];
#pragma unroll
    for (int i = 0; i < 2; ++i)
#pragma unroll
        for (int j = 0; j < 4; ++j) acc[i][j] = zero;

    const int nK = K / BK;

    // A tile stage: 128 rows x 32 k x 2B = 8 KB = 512 chunks of 16 B.
    // chunk c: row = c>>2, kc = c&3. Thread t handles chunks t and t+256.
    // ---- stage buffer 0
    {
#pragma unroll
        for (int i = 0; i < 2; ++i) {
            int c = tid + i * NT;
            int row = c >> 2, kc = c & 3;
            copy16_g2l(H + (size_t)(mBase + row) * LDH + kc * 8,
                       &As[0][row * ROWP + kc * 8]);
        }
#pragma unroll
        for (int i = 0; i < 16; ++i) {
            int e = tid + i * NT;
            int k = e >> 7, n = e & 127;
            float w = wptr[(size_t)k * ldb + (cb + n)];
            Bs[0][n * ROWP + k] = (unsigned short)bf16_bits(w);
        }
    }
    wait_async_copies();
    __syncthreads();

    for (int kt = 0; kt < nK; ++kt) {
        const int cur = kt & 1;
        if (kt + 1 < nK) {
            const int k0 = (kt + 1) * BK;
            unsigned short* Ad = As[cur ^ 1];
            unsigned short* Bp = Bs[cur ^ 1];
#pragma unroll
            for (int i = 0; i < 2; ++i) {
                int c = tid + i * NT;
                int row = c >> 2, kc = c & 3;
                copy16_g2l(H + (size_t)(mBase + row) * LDH + k0 + kc * 8,
                           &Ad[row * ROWP + kc * 8]);
            }
#pragma unroll
            for (int i = 0; i < 16; ++i) {
                int e = tid + i * NT;
                int k = e >> 7, n = e & 127;
                float w = wptr[(size_t)(k0 + k) * ldb + (cb + n)];
                Bp[n * ROWP + k] = (unsigned short)bf16_bits(w);
            }
        }
        const unsigned short* Ap = As[cur];
        const unsigned short* Bp = Bs[cur];
        v16bf a0 = load_frag(Ap, wm,      lane);
        v16bf a1 = load_frag(Ap, wm + 16, lane);
        v16bf b0 = load_frag(Bp, wn,      lane);
        v16bf b1 = load_frag(Bp, wn + 16, lane);
        v16bf b2 = load_frag(Bp, wn + 32, lane);
        v16bf b3 = load_frag(Bp, wn + 48, lane);
        acc[0][0] = WMMA_BF16(a0, b0, acc[0][0]);
        acc[0][1] = WMMA_BF16(a0, b1, acc[0][1]);
        acc[0][2] = WMMA_BF16(a0, b2, acc[0][2]);
        acc[0][3] = WMMA_BF16(a0, b3, acc[0][3]);
        acc[1][0] = WMMA_BF16(a1, b0, acc[1][0]);
        acc[1][1] = WMMA_BF16(a1, b1, acc[1][1]);
        acc[1][2] = WMMA_BF16(a1, b2, acc[1][2]);
        acc[1][3] = WMMA_BF16(a1, b3, acc[1][3]);
        wait_async_copies();
        __syncthreads();
    }

    // ---- epilogue: bias + f32 store
    const int rHalf = (lane >> 4) * 8;
    const int nLoc  = (lane & 15);
#pragma unroll
    for (int i = 0; i < 2; ++i) {
#pragma unroll
        for (int j = 0; j < 4; ++j) {
            int loc  = wn + j * 16 + nLoc;      // 0..127 within block tile
            float bv = bptr[cb + loc];
            int rowB = mBase + wm + i * 16 + rHalf;
#pragma unroll
            for (int r = 0; r < 8; ++r) {
                OUT[(size_t)(rowB + r) * LDO + (nBase + loc)] = acc[i][j][r] + bv;
            }
        }
    }
}

// ---------------------------------------------------------------------------
extern "C" void kernel_launch(void* const* d_in, const int* in_sizes, int n_in,
                              void* d_out, int out_size, void* d_ws, size_t ws_size,
                              hipStream_t stream) {
    const float* x   = (const float*)d_in[0];
    const int*   idx = (const int*)  d_in[1];
    const float* W1  = (const float*)d_in[2];
    const float* b1  = (const float*)d_in[3];
    const float* W2  = (const float*)d_in[4];
    const float* b2  = (const float*)d_in[5];
    const float* We  = (const float*)d_in[6];
    const float* be  = (const float*)d_in[7];
    float* out = (float*)d_out;
    unsigned short* H = (unsigned short*)d_ws;   // 32768*4096 bf16 = 256 MB

    dim3 block(NT);
    dim3 g1(4096 / BN, 32768 / BM);   // 32 x 256
    dim3 g2(1024 / BN, 32768 / BM);   //  8 x 256

    hipLaunchKernelGGL(fc1_gelu_kernel, g1, block, 0, stream, x, W1, b1, H);
    hipLaunchKernelGGL(fc2_moe_kernel,  g2, block, 0, stream, H, W2, b2, We, be, idx, out);
}
